// TransformerBlock_77163382440443
// MI455X (gfx1250) — matrix operand order repaired and checked by reading, not compiled, not run on hardware
//
#include <hip/hip_runtime.h>

// ---------------------------------------------------------------------------
// Types for CDNA5 WMMA (wave32): bf16 16x16x32, fp32 accumulate
// ---------------------------------------------------------------------------
typedef __attribute__((ext_vector_type(16))) __bf16 v16bf;
typedef __attribute__((ext_vector_type(8)))  float  v8f;
typedef __attribute__((ext_vector_type(4)))  unsigned int u32x4;

union FragU { u32x4 u[2]; v16bf v; };

__device__ inline v16bf ld_frag(const __bf16* p0, const __bf16* p1) {
    FragU f;
    f.u[0] = *reinterpret_cast<const u32x4*>(p0);
    f.u[1] = *reinterpret_cast<const u32x4*>(p1);
    return f.v;
}

__device__ inline v8f wmma_bf16(v16bf a, v16bf b, v8f c) {
    return __builtin_amdgcn_wmma_f32_16x16x32_bf16(
        false, a, false, b, (short)0, c, false, false);
}

// ---------------------------------------------------------------------------
// CDNA5 async global->LDS copy (ASYNCcnt-tracked, no VGPR data round trip)
// ---------------------------------------------------------------------------
#define USE_ASYNC_LDS 1

#define LDS_AS __attribute__((address_space(3)))
__device__ inline unsigned lds_offset(const void* p) {
    // addrspace(3) pointers are 32-bit LDS byte addresses on amdgcn
    return (unsigned)(unsigned long long)(LDS_AS const void*)p;
}

__device__ inline void async_ld_b128(unsigned lds_off, const void* g) {
    asm volatile("global_load_async_to_lds_b128 %0, %1, off"
                 :: "v"(lds_off), "v"((unsigned long long)g)
                 : "memory");
}

__device__ inline void wait_asynccnt0() {
#if __has_builtin(__builtin_amdgcn_s_wait_asynccnt)
    __builtin_amdgcn_s_wait_asynccnt(0);
#else
    asm volatile("s_wait_asynccnt 0x0" ::: "memory");
#endif
}

#define D_MODEL 2048
#define D_FF    8192
#define NHEAD   16
#define DK      128
#define SEQ     2048
#define BATCH   2
#define MROWS   (BATCH * SEQ)   // 4096

// ---------------------------------------------------------------------------
// fp32 -> bf16 weight cast
// ---------------------------------------------------------------------------
__global__ void castw_kernel(const float* __restrict__ w, __bf16* __restrict__ o, int n) {
    int i = blockIdx.x * 256 + threadIdx.x;
    if (i < n) o[i] = (__bf16)w[i];
}

// ---------------------------------------------------------------------------
// RMSNorm: one 256-thread block per row of 2048, bf16 output
// ---------------------------------------------------------------------------
__global__ __launch_bounds__(256) void rmsnorm_kernel(const float* __restrict__ x,
                                                      const float* __restrict__ g,
                                                      __bf16* __restrict__ out) {
    __shared__ float red[8];
    const int row = blockIdx.x;
    const float* xr = x + (size_t)row * D_MODEL;
    float ss = 0.f;
    for (int i = threadIdx.x; i < D_MODEL; i += 256) { float v = xr[i]; ss += v * v; }
    for (int off = 16; off > 0; off >>= 1) ss += __shfl_xor(ss, off, 32);
    const int wv = threadIdx.x >> 5, lane = threadIdx.x & 31;
    if (lane == 0) red[wv] = ss;
    __syncthreads();
    if (wv == 0) {
        float v = (lane < 8) ? red[lane] : 0.f;
        for (int off = 4; off > 0; off >>= 1) v += __shfl_xor(v, off, 32);
        if (lane == 0) red[0] = rsqrtf(v / (float)D_MODEL + 1e-8f);
    }
    __syncthreads();
    const float r = red[0];
    for (int i = threadIdx.x; i < D_MODEL; i += 256)
        out[(size_t)row * D_MODEL + i] = (__bf16)(xr[i] * r * g[i]);
}

// ---------------------------------------------------------------------------
// RoPE + relayout: fp32 [b,s,h*dk] -> bf16 [b*h, s, dk]
// ---------------------------------------------------------------------------
__global__ __launch_bounds__(256) void rope_cast_kernel(const float* __restrict__ src,
                                                        const int* __restrict__ pos,
                                                        __bf16* __restrict__ dst) {
    const int row = blockIdx.x;              // b*SEQ + s
    const int b = row >> 11, s = row & (SEQ - 1);
    const float p = (float)pos[s];
    for (int i = threadIdx.x; i < NHEAD * (DK / 2); i += 256) {
        const int h = i >> 6, j = i & 63;    // head, pair index
        const float inv = __expf(-9.210340371976184f * (2.f * (float)j) / (float)DK);
        float sn, cs;
        __sincosf(p * inv, &sn, &cs);
        const float xe = src[(size_t)row * D_MODEL + h * DK + 2 * j];
        const float xo = src[(size_t)row * D_MODEL + h * DK + 2 * j + 1];
        const size_t o = ((size_t)(b * NHEAD + h) * SEQ + s) * DK + 2 * j;
        dst[o]     = (__bf16)(xe * cs - xo * sn);
        dst[o + 1] = (__bf16)(xe * sn + xo * cs);
    }
}

// V relayout: fp32 [b,s,h*dk] -> bf16 [b*h, dk, s]  (transposed for PV fragments)
__global__ __launch_bounds__(256) void vcast_kernel(const float* __restrict__ V,
                                                    __bf16* __restrict__ Vt) {
    const size_t i = (size_t)blockIdx.x * 256 + threadIdx.x;   // over b*s*d
    const size_t row = i >> 11;           // b*SEQ+s
    const int col = (int)(i & (D_MODEL - 1));
    const int b = (int)(row >> 11), s = (int)(row & (SEQ - 1));
    const int h = col >> 7, dkc = col & (DK - 1);
    Vt[((size_t)(b * NHEAD + h) * DK + dkc) * SEQ + s] = (__bf16)V[i];
}

// ---------------------------------------------------------------------------
// NT GEMM: C(MxN) = A(MxK) * B(NxK)^T, bf16 in, fp32 acc via WMMA.
// 128x128 block tile, BK=32, 8 waves, each wave: 2x4 of 16x16 WMMA tiles.
// Tiles staged in LDS via GLOBAL_LOAD_ASYNC_TO_LDS_B128 (ASYNCcnt).
// EPI: 0 = store f32, 2 = store f32 + residual R, 3 = store bf16 silu(R)*acc
// ---------------------------------------------------------------------------
template <int EPI>
__global__ __launch_bounds__(256) void gemm_bf16_kernel(const __bf16* __restrict__ A,
                                                        const __bf16* __restrict__ B,
                                                        float* __restrict__ Cf,
                                                        __bf16* __restrict__ Cb,
                                                        const float* __restrict__ R,
                                                        int M, int N, int K) {
    __shared__ __bf16 As[128 * 32];
    __shared__ __bf16 Bs[128 * 32];
    const int tid = threadIdx.x;
    const int lane = tid & 31, wv = tid >> 5;
    const int wm = wv & 3, wn = wv >> 2;                 // 4 waves along M, 2 along N
    const int m0 = blockIdx.y * 128, n0 = blockIdx.x * 128;
    const int hl = lane >> 4, l16 = lane & 15;

    v8f acc[2][4];
    for (int i = 0; i < 2; i++)
        for (int j = 0; j < 4; j++)
            for (int e = 0; e < 8; e++) acc[i][j][e] = 0.f;

    const int lrow = tid >> 1;            // 0..127
    const int lcol = (tid & 1) << 4;      // 0 or 16
    const unsigned la0 = lds_offset(&As[lrow * 32 + lcol]);
    const unsigned la1 = lds_offset(&As[lrow * 32 + lcol + 8]);
    const unsigned lb0 = lds_offset(&Bs[lrow * 32 + lcol]);
    const unsigned lb1 = lds_offset(&Bs[lrow * 32 + lcol + 8]);

    for (int k0 = 0; k0 < K; k0 += 32) {
        const __bf16* ap = A + (size_t)(m0 + lrow) * K + k0 + lcol;
        const __bf16* bp = B + (size_t)(n0 + lrow) * K + k0 + lcol;
#if USE_ASYNC_LDS
        __syncthreads();                 // WAR: others done reading previous tile
        async_ld_b128(la0, ap);
        async_ld_b128(la1, ap + 8);
        async_ld_b128(lb0, bp);
        async_ld_b128(lb1, bp + 8);
        wait_asynccnt0();
        __syncthreads();                 // visibility of async-written LDS
#else
        const u32x4* ag = reinterpret_cast<const u32x4*>(ap);
        u32x4 a0 = ag[0], a1 = ag[1];
        const u32x4* bg = reinterpret_cast<const u32x4*>(bp);
        u32x4 b0 = bg[0], b1 = bg[1];
        __syncthreads();
        *reinterpret_cast<u32x4*>(&As[lrow * 32 + lcol])     = a0;
        *reinterpret_cast<u32x4*>(&As[lrow * 32 + lcol + 8]) = a1;
        *reinterpret_cast<u32x4*>(&Bs[lrow * 32 + lcol])     = b0;
        *reinterpret_cast<u32x4*>(&Bs[lrow * 32 + lcol + 8]) = b1;
        __syncthreads();
#endif

        v16bf af[2], bf[4];
#pragma unroll
        for (int sm = 0; sm < 2; sm++) {
            const int r = wm * 32 + sm * 16 + l16;
            af[sm] = ld_frag(&As[r * 32 + hl * 8], &As[r * 32 + hl * 8 + 16]);
        }
#pragma unroll
        for (int sn = 0; sn < 4; sn++) {
            const int r = wn * 64 + sn * 16 + l16;
            bf[sn] = ld_frag(&Bs[r * 32 + hl * 8], &Bs[r * 32 + hl * 8 + 16]);
        }
#pragma unroll
        for (int sm = 0; sm < 2; sm++)
#pragma unroll
            for (int sn = 0; sn < 4; sn++)
                acc[sm][sn] = wmma_bf16(af[sm], bf[sn], acc[sm][sn]);
    }

#pragma unroll
    for (int sm = 0; sm < 2; sm++) {
#pragma unroll
        for (int sn = 0; sn < 4; sn++) {
            const int colg = n0 + wn * 64 + sn * 16 + l16;
#pragma unroll
            for (int r = 0; r < 8; r++) {
                const int rowg = m0 + wm * 32 + sm * 16 + hl * 8 + r;
                const size_t idx = (size_t)rowg * N + colg;
                const float v = acc[sm][sn][r];
                if (EPI == 0) {
                    Cf[idx] = v;
                } else if (EPI == 2) {
                    Cf[idx] = v + R[idx];
                } else {
                    const float a = R[idx];
                    const float s = a / (1.f + __expf(-a));
                    Cb[idx] = (__bf16)(s * v);
                }
            }
        }
    }
}

// ---------------------------------------------------------------------------
// Flash attention (causal): one wave per 16-query tile per (b,h).
// Q,K: bf16 [b*h, S, DK]; V: bf16 [b*h, DK, S]; out ctx: bf16 [b, S, h*DK]
// ---------------------------------------------------------------------------
__global__ __launch_bounds__(256) void flash_attn_kernel(const __bf16* __restrict__ Q,
                                                         const __bf16* __restrict__ Kk,
                                                         const __bf16* __restrict__ Vt,
                                                         __bf16* __restrict__ ctx) {
    __shared__ __bf16 Pl[8][16 * 32];
    const int tid = threadIdx.x, lane = tid & 31, wv = tid >> 5;
    const int hl = lane >> 4, l16 = lane & 15;
    const int gw = blockIdx.x * 8 + wv;
    const int bh = gw >> 7;                 // (b*NHEAD+h)
    const int qt = gw & 127;
    const int q0 = qt * 16;
    const __bf16* Qb = Q  + (size_t)bh * SEQ * DK;
    const __bf16* Kb = Kk + (size_t)bh * SEQ * DK;
    const __bf16* Vb = Vt + (size_t)bh * DK * SEQ;
    const float scale = 0.08838834764831845f;   // 1/sqrt(128)

    v16bf qf[4];
#pragma unroll
    for (int f = 0; f < 4; f++) {
        const __bf16* p = Qb + (size_t)(q0 + l16) * DK + f * 32 + hl * 8;
        qf[f] = ld_frag(p, p + 16);
    }

    float m[8], l[8];
    v8f o[8];
    for (int r = 0; r < 8; r++) { m[r] = -1e30f; l[r] = 0.f; }
    for (int t = 0; t < 8; t++)
        for (int r = 0; r < 8; r++) o[t][r] = 0.f;

    const int nkb = (q0 + 16 + 31) >> 5;    // causal: key blocks of 32
    for (int kb = 0; kb < nkb; kb++) {
        const int k0 = kb * 32;
        v8f s0, s1;
        for (int r = 0; r < 8; r++) { s0[r] = 0.f; s1[r] = 0.f; }
#pragma unroll
        for (int f = 0; f < 4; f++) {
            const __bf16* p0 = Kb + (size_t)(k0 + l16) * DK + f * 32 + hl * 8;
            s0 = wmma_bf16(qf[f], ld_frag(p0, p0 + 16), s0);
            const __bf16* p1 = Kb + (size_t)(k0 + 16 + l16) * DK + f * 32 + hl * 8;
            s1 = wmma_bf16(qf[f], ld_frag(p1, p1 + 16), s1);
        }
#pragma unroll
        for (int r = 0; r < 8; r++) {
            const int rowq = q0 + hl * 8 + r;
            float e0 = s0[r] * scale, e1 = s1[r] * scale;
            if (k0 + l16 > rowq)      e0 = -1e30f;
            if (k0 + 16 + l16 > rowq) e1 = -1e30f;
            float t = fmaxf(e0, e1);
            for (int off = 1; off < 16; off <<= 1) t = fmaxf(t, __shfl_xor(t, off, 32));
            const float mn = fmaxf(m[r], t);
            const float alpha = __expf(m[r] - mn);
            m[r] = mn;
            const float p0v = __expf(e0 - mn), p1v = __expf(e1 - mn);
            float rs = p0v + p1v;
            for (int off = 1; off < 16; off <<= 1) rs += __shfl_xor(rs, off, 32);
            l[r] = l[r] * alpha + rs;
#pragma unroll
            for (int t8 = 0; t8 < 8; t8++) o[t8][r] *= alpha;
            __bf16* pl = &Pl[wv][(hl * 8 + r) * 32];
            pl[l16]      = (__bf16)p0v;
            pl[16 + l16] = (__bf16)p1v;
        }
        // P (16x32) back out of LDS in A-fragment layout; DS ops in-order per wave
        const __bf16* pb = &Pl[wv][l16 * 32 + hl * 8];
        const v16bf pf = ld_frag(pb, pb + 16);
#pragma unroll
        for (int t8 = 0; t8 < 8; t8++) {
            const __bf16* vp = Vb + (size_t)(t8 * 16 + l16) * SEQ + k0 + hl * 8;
            o[t8] = wmma_bf16(pf, ld_frag(vp, vp + 16), o[t8]);
        }
    }

    const int b = bh >> 4, hd = bh & 15;
#pragma unroll
    for (int t8 = 0; t8 < 8; t8++) {
        const int colc = hd * DK + t8 * 16 + l16;
#pragma unroll
        for (int r = 0; r < 8; r++) {
            const int rowq = q0 + hl * 8 + r;
            ctx[((size_t)(b * SEQ + rowq)) * D_MODEL + colc] = (__bf16)(o[t8][r] / l[r]);
        }
    }
}

// ---------------------------------------------------------------------------
// Host orchestration
// ---------------------------------------------------------------------------
extern "C" void kernel_launch(void* const* d_in, const int* in_sizes, int n_in,
                              void* d_out, int out_size, void* d_ws, size_t ws_size,
                              hipStream_t stream) {
    (void)in_sizes; (void)n_in; (void)out_size; (void)ws_size;
    const float* x   = (const float*)d_in[0];
    const float* wq  = (const float*)d_in[1];
    const float* wk  = (const float*)d_in[2];
    const float* wvp = (const float*)d_in[3];
    const float* wo  = (const float*)d_in[4];
    const float* g1  = (const float*)d_in[5];
    const float* g2  = (const float*)d_in[6];
    const float* w1  = (const float*)d_in[7];
    const float* w3  = (const float*)d_in[8];
    const float* w2  = (const float*)d_in[9];
    const int*   pos = (const int*)d_in[10];
    float* out = (float*)d_out;

    char* ws = (char*)d_ws;
    size_t off = 0;
    auto alloc = [&](size_t bytes) -> void* {
        void* p = ws + off;
        off += (bytes + 255) & ~(size_t)255;
        return p;
    };

    const size_t DD = (size_t)D_MODEL * D_MODEL;     // 4M
    const size_t FD = (size_t)D_FF * D_MODEL;        // 16.8M
    const size_t MD = (size_t)MROWS * D_MODEL;       // 8.4M
    const size_t MF = (size_t)MROWS * D_FF;          // 33.6M

    __bf16* wq_b = (__bf16*)alloc(DD * 2);
    __bf16* wk_b = (__bf16*)alloc(DD * 2);
    __bf16* wv_b = (__bf16*)alloc(DD * 2);
    __bf16* wo_b = (__bf16*)alloc(DD * 2);
    __bf16* w1_b = (__bf16*)alloc(FD * 2);
    __bf16* w3_b = (__bf16*)alloc(FD * 2);
    __bf16* w2_b = (__bf16*)alloc(FD * 2);
    __bf16* xn_b  = (__bf16*)alloc(MD * 2);
    float*  Qf    = (float*) alloc(MD * 4);
    float*  Kf    = (float*) alloc(MD * 4);
    float*  Vf    = (float*) alloc(MD * 4);
    __bf16* Q_b   = (__bf16*)alloc(MD * 2);
    __bf16* K_b   = (__bf16*)alloc(MD * 2);
    __bf16* Vt_b  = (__bf16*)alloc(MD * 2);
    __bf16* ctx_b = (__bf16*)alloc(MD * 2);
    float*  x1f   = (float*) alloc(MD * 4);
    __bf16* xn2_b = (__bf16*)alloc(MD * 2);
    float*  h1f   = (float*) alloc(MF * 4);
    __bf16* hh_b  = (__bf16*)alloc(MF * 2);

    // 1. weight casts
    castw_kernel<<<(int)((DD + 255) / 256), 256, 0, stream>>>(wq,  wq_b, (int)DD);
    castw_kernel<<<(int)((DD + 255) / 256), 256, 0, stream>>>(wk,  wk_b, (int)DD);
    castw_kernel<<<(int)((DD + 255) / 256), 256, 0, stream>>>(wvp, wv_b, (int)DD);
    castw_kernel<<<(int)((DD + 255) / 256), 256, 0, stream>>>(wo,  wo_b, (int)DD);
    castw_kernel<<<(int)((FD + 255) / 256), 256, 0, stream>>>(w1,  w1_b, (int)FD);
    castw_kernel<<<(int)((FD + 255) / 256), 256, 0, stream>>>(w3,  w3_b, (int)FD);
    castw_kernel<<<(int)((FD + 255) / 256), 256, 0, stream>>>(w2,  w2_b, (int)FD);

    // 2. rmsnorm(x, g1) -> bf16
    rmsnorm_kernel<<<MROWS, 256, 0, stream>>>(x, g1, xn_b);

    // 3. Q/K/V projections (NT GEMM, fp32 out)
    dim3 gqkv(D_MODEL / 128, MROWS / 128);
    gemm_bf16_kernel<0><<<gqkv, 256, 0, stream>>>(xn_b, wq_b, Qf, nullptr, nullptr, MROWS, D_MODEL, D_MODEL);
    gemm_bf16_kernel<0><<<gqkv, 256, 0, stream>>>(xn_b, wk_b, Kf, nullptr, nullptr, MROWS, D_MODEL, D_MODEL);
    gemm_bf16_kernel<0><<<gqkv, 256, 0, stream>>>(xn_b, wv_b, Vf, nullptr, nullptr, MROWS, D_MODEL, D_MODEL);

    // 4. RoPE + relayout
    rope_cast_kernel<<<MROWS, 256, 0, stream>>>(Qf, pos, Q_b);
    rope_cast_kernel<<<MROWS, 256, 0, stream>>>(Kf, pos, K_b);
    vcast_kernel<<<(int)(MD / 256), 256, 0, stream>>>(Vf, Vt_b);

    // 5. flash attention -> ctx bf16
    flash_attn_kernel<<<(BATCH * NHEAD * (SEQ / 16)) / 8, 256, 0, stream>>>(Q_b, K_b, Vt_b, ctx_b);

    // 6. x1 = x + ctx @ wo^T
    gemm_bf16_kernel<2><<<gqkv, 256, 0, stream>>>(ctx_b, wo_b, x1f, nullptr, x, MROWS, D_MODEL, D_MODEL);

    // 7. rmsnorm(x1, g2)
    rmsnorm_kernel<<<MROWS, 256, 0, stream>>>(x1f, g2, xn2_b);

    // 8. h1 = xn2 @ w1^T (fp32)
    dim3 gff(D_FF / 128, MROWS / 128);
    gemm_bf16_kernel<0><<<gff, 256, 0, stream>>>(xn2_b, w1_b, h1f, nullptr, nullptr, MROWS, D_FF, D_MODEL);

    // 9. hh = silu(h1) * (xn2 @ w3^T)  (bf16)
    gemm_bf16_kernel<3><<<gff, 256, 0, stream>>>(xn2_b, w3_b, nullptr, hh_b, h1f, MROWS, D_FF, D_MODEL);

    // 10. out = x1 + hh @ w2^T
    gemm_bf16_kernel<2><<<gqkv, 256, 0, stream>>>(hh_b, w2_b, out, nullptr, x1f, MROWS, D_MODEL, D_FF);
}
